// HardOpBank_71347996721421
// MI455X (gfx1250) — matrix operand order repaired and checked by reading, not compiled, not run on hardware
//
#include <hip/hip_runtime.h>
#include <hip/hip_bf16.h>

#define N_OPS 8
#define DIMK  512
#define MTILE 64          // tokens per block (4 WMMA sub-tiles of 16)
#define XPAD  520         // 512 + 8 bf16 pad -> row stride 1040B, 16B-aligned, bank-skewed

typedef __attribute__((ext_vector_type(16))) __bf16 v16bf;
typedef __attribute__((ext_vector_type(8)))  __bf16 v8bf;
typedef __attribute__((ext_vector_type(4)))  __bf16 v4bf;
typedef __attribute__((ext_vector_type(8)))  float  v8f;
typedef __attribute__((ext_vector_type(4)))  float  v4f;

// ---------------------------------------------------------------------------
// Activations (match jax: exact-erf gelu, relu, relu^2, silu, tanh, sigmoid,
// identity, negate)
// ---------------------------------------------------------------------------
__device__ __forceinline__ float apply_act(int e, float h) {
    switch (e & 7) {
        case 0:  return 0.5f * h * (1.0f + erff(h * 0.70710678118654752440f));
        case 1:  return fmaxf(h, 0.0f);
        case 2:  { float t = fmaxf(h, 0.0f); return t * t; }
        case 3:  return h / (1.0f + expf(-h));          // silu
        case 4:  return tanhf(h);
        case 5:  return 1.0f / (1.0f + expf(-h));       // sigmoid
        case 6:  return h;
        default: return -h;
    }
}

// ---------------------------------------------------------------------------
// Kernel 0: zero the 8 per-expert counters in workspace
// ---------------------------------------------------------------------------
__global__ void zero_counts_kernel(int* __restrict__ counts) {
    if (threadIdx.x < N_OPS) counts[threadIdx.x] = 0;
}

// ---------------------------------------------------------------------------
// Kernel 1: per-token routing. idx = argmax(logits + gumbel) (first-max wins,
// matching jnp.argmax). Append token id into its expert bucket.
// ---------------------------------------------------------------------------
__global__ __launch_bounds__(256)
void route_kernel(const float* __restrict__ logits,
                  const float* __restrict__ gumbel,
                  int* __restrict__ counts,
                  int* __restrict__ bucket,
                  int n_tok) {
    int t = blockIdx.x * blockDim.x + threadIdx.x;
    if (t >= n_tok) return;
    float best = -3.402823466e+38f;
    int bi = 0;
    #pragma unroll
    for (int i = 0; i < N_OPS; ++i) {
        float s = logits[t * N_OPS + i] + gumbel[t * N_OPS + i];
        if (s > best) { best = s; bi = i; }
    }
    int pos = atomicAdd(&counts[bi], 1);
    bucket[(size_t)bi * n_tok + pos] = t;
}

// ---------------------------------------------------------------------------
// Kernel 2: pre-convert W (fp32) into bf16 hi/lo planes in workspace.
// Row-major layout preserved; keeps the GEMM inner loop free of conversion
// VALU and halves B-operand L2 bytes.
// ---------------------------------------------------------------------------
__global__ __launch_bounds__(256)
void wconv_kernel(const float* __restrict__ W,
                  __bf16* __restrict__ Whi,
                  __bf16* __restrict__ Wlo,
                  int n) {
    int i = (blockIdx.x * 256 + threadIdx.x) * 4;
    if (i >= n) return;
    v4f v = *(const v4f*)(W + i);
    v4bf h4, l4;
    #pragma unroll
    for (int u = 0; u < 4; ++u) {
        float  f = v[u];
        __bf16 h = (__bf16)f;
        h4[u] = h;
        l4[u] = (__bf16)(f - (float)h);
    }
    *(v4bf*)(Whi + i) = h4;    // 8B store, aligned (i % 4 == 0)
    *(v4bf*)(Wlo + i) = l4;
}

// ---------------------------------------------------------------------------
// Kernel 3: grouped GEMM + activation, bf16 hi/lo split on WMMA.
// Block = 256 threads (8 waves of 32). One block = 64 tokens of one expert
// (4 M sub-tiles), full 512 outputs. Wave w owns N-tiles {4w..4w+3}.
// Per (kc, nt): 4x global_load_b128 (B hi/lo frags) amortized over 12 WMMAs
// (4 M sub-tiles x 3 hi/lo-split terms) -> ~96 FLOP per L2 byte.
// ---------------------------------------------------------------------------
__global__ __launch_bounds__(256)
void gemm_kernel(const float*  __restrict__ x,
                 const __bf16* __restrict__ Whi,
                 const __bf16* __restrict__ Wlo,
                 const int*    __restrict__ counts,
                 const int*    __restrict__ bucket,
                 float*        __restrict__ out,
                 int n_tok, int tiles_per_expert) {
    const int e    = blockIdx.x / tiles_per_expert;
    const int j    = blockIdx.x - e * tiles_per_expert;
    const int cnt  = counts[e];
    const int base = j * MTILE;
    if (base >= cnt) return;

    // Dynamic LDS: [sXhi 64x520 bf16][sXlo 64x520 bf16][s_tok 64 int] ~ 130.3 KB
    extern __shared__ __bf16 smem[];
    __bf16 (*sXhi)[XPAD] = (__bf16(*)[XPAD])smem;
    __bf16 (*sXlo)[XPAD] = ((__bf16(*)[XPAD])smem) + MTILE;
    int* s_tok = (int*)(smem + 2 * MTILE * XPAD);

    const int tid = threadIdx.x;
    if (tid < MTILE) {
        int idx = base + tid;
        s_tok[tid] = (idx < cnt) ? bucket[(size_t)e * n_tok + idx] : -1;
    }
    __syncthreads();

    // ---- Stage x tile as bf16 hi/lo. Thread covers 128 contiguous cols of one row.
    {
        const int row = tid >> 2;            // 0..63
        const int cb  = (tid & 3) * 128;     // column base
        const int tok = s_tok[row];
        #pragma unroll 4
        for (int q = 0; q < 128; q += 4) {
            v4f v;
            if (tok >= 0) v = *(const v4f*)(x + (size_t)tok * DIMK + cb + q);
            else          v = (v4f){0.f, 0.f, 0.f, 0.f};
            v4bf h4, l4;
            #pragma unroll
            for (int u = 0; u < 4; ++u) {
                float  f = v[u];
                __bf16 h = (__bf16)f;
                h4[u] = h;
                l4[u] = (__bf16)(f - (float)h);
            }
            *(v4bf*)&sXhi[row][cb + q] = h4;   // 8B ds stores, aligned
            *(v4bf*)&sXlo[row][cb + q] = l4;
        }
    }
    __syncthreads();

    const int wave = tid >> 5;
    const int lane = tid & 31;
    const int half = lane >> 4;      // which 16-lane half
    const int l15  = lane & 15;

    v8f acc[4][4];                   // [mt][nt]
    #pragma unroll
    for (int mt = 0; mt < 4; ++mt)
        #pragma unroll
        for (int nt = 0; nt < 4; ++nt)
            #pragma unroll
            for (int q = 0; q < 8; ++q) acc[mt][nt][q] = 0.0f;

    const __bf16* Whi_e = Whi + (size_t)e * DIMK * DIMK;
    const __bf16* Wlo_e = Wlo + (size_t)e * DIMK * DIMK;
    const int koffA = half * 8;      // A frag: lanes 0-15 K=0..7 & 16..23, lanes 16-31 K=8..15 & 24..31
    const int koffB = half * 8;      // B frag: lanes 0-15 K=0..7 & 16..23, lanes 16-31 K=8..15 & 24..31

    for (int kc = 0; kc < DIMK; kc += 32) {
        v16bf ahi[4], alo[4];
        #pragma unroll
        for (int mt = 0; mt < 4; ++mt) {
            const int m = mt * 16 + l15;
            v8bf a0h = *(const v8bf*)&sXhi[m][kc + koffA];
            v8bf a1h = *(const v8bf*)&sXhi[m][kc + koffA + 16];
            v8bf a0l = *(const v8bf*)&sXlo[m][kc + koffA];
            v8bf a1l = *(const v8bf*)&sXlo[m][kc + koffA + 16];
            ahi[mt] = __builtin_shufflevector(a0h, a1h, 0,1,2,3,4,5,6,7,8,9,10,11,12,13,14,15);
            alo[mt] = __builtin_shufflevector(a0l, a1l, 0,1,2,3,4,5,6,7,8,9,10,11,12,13,14,15);
        }

        #pragma unroll
        for (int nt = 0; nt < 4; ++nt) {
            const int o = (wave * 4 + nt) * 16 + l15;        // output feature = B column
            const size_t wo = (size_t)o * DIMK + kc + koffB; // K runs koffB..+7 and koffB+16..+23
            v8bf bh0 = *(const v8bf*)(Whi_e + wo);
            v8bf bh1 = *(const v8bf*)(Whi_e + wo + 16);
            v8bf bl0 = *(const v8bf*)(Wlo_e + wo);
            v8bf bl1 = *(const v8bf*)(Wlo_e + wo + 16);
            v16bf bhi = __builtin_shufflevector(bh0, bh1, 0,1,2,3,4,5,6,7,8,9,10,11,12,13,14,15);
            v16bf blo = __builtin_shufflevector(bl0, bl1, 0,1,2,3,4,5,6,7,8,9,10,11,12,13,14,15);

            #pragma unroll
            for (int mt = 0; mt < 4; ++mt) {
                // acc += xh*wh + xl*wh + xh*wl  (hi/lo split ~= fp32 product)
                acc[mt][nt] = __builtin_amdgcn_wmma_f32_16x16x32_bf16(
                    false, ahi[mt], false, bhi, (short)0, acc[mt][nt], false, false);
                acc[mt][nt] = __builtin_amdgcn_wmma_f32_16x16x32_bf16(
                    false, alo[mt], false, bhi, (short)0, acc[mt][nt], false, false);
                acc[mt][nt] = __builtin_amdgcn_wmma_f32_16x16x32_bf16(
                    false, ahi[mt], false, blo, (short)0, acc[mt][nt], false, false);
            }
        }
    }

    // ---- Epilogue: activation + masked scatter.
    // C/D layout: VGPR r -> M = mt*16 + r + 8*half, N = l15.
    #pragma unroll
    for (int mt = 0; mt < 4; ++mt) {
        #pragma unroll
        for (int nt = 0; nt < 4; ++nt) {
            const int o = (wave * 4 + nt) * 16 + l15;
            #pragma unroll
            for (int r = 0; r < 8; ++r) {
                const int m   = mt * 16 + r + half * 8;
                const int tok = s_tok[m];
                if (tok >= 0)
                    out[(size_t)tok * DIMK + o] = apply_act(e, acc[mt][nt][r]);
            }
        }
    }
}

// ---------------------------------------------------------------------------
// Fallback (only if workspace too small): per-token VALU matvec, still correct.
// ---------------------------------------------------------------------------
__global__ __launch_bounds__(256)
void fallback_kernel(const float* __restrict__ x,
                     const float* __restrict__ logits,
                     const float* __restrict__ gumbel,
                     const float* __restrict__ W,
                     float* __restrict__ out) {
    const int t   = blockIdx.x;
    const int tid = threadIdx.x;
    __shared__ float sx[DIMK];
    __shared__ int   sIdx;
    if (tid == 0) {
        float best = -3.402823466e+38f; int bi = 0;
        #pragma unroll
        for (int i = 0; i < N_OPS; ++i) {
            float s = logits[t * N_OPS + i] + gumbel[t * N_OPS + i];
            if (s > best) { best = s; bi = i; }
        }
        sIdx = bi;
    }
    sx[tid]       = x[(size_t)t * DIMK + tid];
    sx[tid + 256] = x[(size_t)t * DIMK + tid + 256];
    __syncthreads();
    const int e = sIdx;
    for (int o = tid; o < DIMK; o += 256) {
        const float* wr = W + ((size_t)e * DIMK + o) * DIMK;
        float acc = 0.0f;
        for (int d = 0; d < DIMK; ++d) acc = fmaf(sx[d], wr[d], acc);
        out[(size_t)t * DIMK + o] = apply_act(e, acc);
    }
}

// ---------------------------------------------------------------------------
extern "C" void kernel_launch(void* const* d_in, const int* in_sizes, int n_in,
                              void* d_out, int out_size, void* d_ws, size_t ws_size,
                              hipStream_t stream) {
    const float* x      = (const float*)d_in[0];
    const float* logits = (const float*)d_in[1];
    const float* gumbel = (const float*)d_in[2];
    const float* W      = (const float*)d_in[3];
    float*       out    = (float*)d_out;

    const int n_tok = in_sizes[1] / N_OPS;           // B*T = 32768
    const int wn    = in_sizes[3];                   // 8*512*512 = 2097152

    // Workspace layout: [counts(8 int)] [bucket(8*n_tok int)] [Whi bf16] [Wlo bf16]
    size_t off_bucket = sizeof(int) * N_OPS;
    size_t off_whi    = (off_bucket + sizeof(int) * (size_t)N_OPS * n_tok + 63) & ~(size_t)63;
    size_t off_wlo    = off_whi + sizeof(__bf16) * (size_t)wn;
    size_t need       = off_wlo + sizeof(__bf16) * (size_t)wn;

    if (ws_size < need) {
        fallback_kernel<<<n_tok, 256, 0, stream>>>(x, logits, gumbel, W, out);
        return;
    }

    int*    counts = (int*)d_ws;
    int*    bucket = (int*)((char*)d_ws + off_bucket);
    __bf16* Whi    = (__bf16*)((char*)d_ws + off_whi);
    __bf16* Wlo    = (__bf16*)((char*)d_ws + off_wlo);
    const int tiles = (n_tok + MTILE - 1) / MTILE;
    const size_t shmem = (size_t)2 * MTILE * XPAD * sizeof(__bf16) + MTILE * sizeof(int);

    zero_counts_kernel<<<1, N_OPS, 0, stream>>>(counts);
    wconv_kernel<<<wn / (256 * 4), 256, 0, stream>>>(W, Whi, Wlo, wn);
    route_kernel<<<(n_tok + 255) / 256, 256, 0, stream>>>(logits, gumbel, counts,
                                                          bucket, n_tok);
    gemm_kernel<<<N_OPS * tiles, 256, shmem, stream>>>(x, Whi, Wlo, counts, bucket,
                                                       out, n_tok, tiles);
}
